// SSMLayer_12893491823416
// MI455X (gfx1250) — matrix operand order repaired and checked by reading, not compiled, not run on hardware
//
#include <hip/hip_runtime.h>
#include <hip/hip_bf16.h>
#include <stdint.h>

#define STATE  256
#define BATCH  16
#define SEQLEN 4096
#define ROWS   (BATCH * SEQLEN)   // 65536 flattened (b,s) rows
#define APITCH 272                // 256 + 16 pad: keeps 32B row alignment, shifts LDS banks

typedef __bf16 bf16;
typedef bf16  v16bf __attribute__((ext_vector_type(16)));
typedef bf16  v8bf  __attribute__((ext_vector_type(8)));
typedef float v8f   __attribute__((ext_vector_type(8)));
typedef unsigned int u32x4 __attribute__((ext_vector_type(4)));
typedef int i32x4 __attribute__((ext_vector_type(4)));
typedef int i32x8 __attribute__((ext_vector_type(8)));

// ---- WMMA helper: D(16x16 f32) += A(16x32 bf16) * B(32x16 bf16) ----
static __device__ __forceinline__ v8f wmma_bf16(v16bf a, v16bf b, v8f c) {
  return __builtin_amdgcn_wmma_f32_16x16x32_bf16(
      /*neg_a=*/false, a, /*neg_b=*/false, b,
      /*c_mod=*/(short)0, c, /*reuse_a=*/false, /*reuse_b=*/false);
}

// A-operand (16x32) lane packing:
// lane-group g (hi): elems 0..15 -> K = kb + 16g .. kb + 16g + 15
static __device__ __forceinline__ v16bf pack_a_f32(const float* row, int kb, int hi) {
  v8f c0 = *(const v8f*)(row + kb + hi * 16);
  v8f c1 = *(const v8f*)(row + kb + hi * 16 + 8);
  v16bf a;
#pragma unroll
  for (int i = 0; i < 8; ++i) { a[i] = (bf16)c0[i]; a[i + 8] = (bf16)c1[i]; }
  return a;
}
static __device__ __forceinline__ v16bf pack_a_bf16(const bf16* row, int kb, int hi) {
  v8bf c0 = *(const v8bf*)(row + kb + hi * 16);
  v8bf c1 = *(const v8bf*)(row + kb + hi * 16 + 8);
  v16bf a;
#pragma unroll
  for (int i = 0; i < 8; ++i) { a[i] = c0[i]; a[i + 8] = c1[i]; }
  return a;
}

// ---- TDM: DMA one u timestep tile u[0:16, t, 0:256] (bf16) into LDS ----
// 2D descriptor: x = state dim (256 contiguous), y = batch (stride SEQ*STATE).
static __device__ __forceinline__ void tdm_load_u_tile(const bf16* u, int t,
                                                       unsigned lds_off) {
  unsigned long long ga =
      (unsigned long long)(uintptr_t)u + (unsigned long long)t * (STATE * 2);
  u32x4 g0;
  g0[0] = 1u;                                   // count=1 (valid), user mode
  g0[1] = lds_off;                              // lds_addr (bytes)
  g0[2] = (unsigned)(ga & 0xFFFFFFFFu);         // global_addr lo
  g0[3] = (unsigned)((ga >> 32) & 0x1FFFFFFu)   // global_addr hi (57-bit)
          | (2u << 30);                         // type = 2 ("image")
  i32x8 g1;
  g1[0] = 0x10000;                              // data_size=1 (2 bytes), mask=0
  g1[1] = (int)(256u << 16);                    // tensor_dim0 = 256 (lo16)
  g1[2] = (int)(16u << 16);                     // dim0 hi=0 | tensor_dim1 = 16
  g1[3] = (int)(256u << 16);                    // dim1 hi=0 | tile_dim0 = 256
  g1[4] = 16;                                   // tile_dim1 = 16, tile_dim2 = 0
  g1[5] = SEQLEN * STATE;                       // tensor_dim0_stride lo32
  g1[6] = 0;                                    // stride hi / dim1_stride lo
  g1[7] = 0;
  i32x4 z4 = {0, 0, 0, 0};
#if __clang_major__ >= 23
  i32x8 z8 = {0, 0, 0, 0, 0, 0, 0, 0};
  __builtin_amdgcn_tensor_load_to_lds(g0, g1, z4, z4, z8, 0);
#else
  __builtin_amdgcn_tensor_load_to_lds(g0, g1, z4, z4, 0);
#endif
}

// ---------------- K1: convert B, C weights to bf16 ----------------
__global__ void ssm_cvt_kernel(const float* __restrict__ B, const float* __restrict__ C,
                               bf16* __restrict__ Bbf, bf16* __restrict__ Cbf) {
  int i = blockIdx.x * blockDim.x + threadIdx.x;
  if (i < STATE * STATE) { Bbf[i] = (bf16)B[i]; Cbf[i] = (bf16)C[i]; }
}

// ---------------- K2: u = x @ B^T  (bf16 out) ----------------
__global__ void ssm_gemm_u_kernel(const float* __restrict__ x,
                                  const bf16* __restrict__ Bbf,
                                  bf16* __restrict__ u) {
  int wave = threadIdx.x >> 5, lane = threadIdx.x & 31;
  int lr = lane & 15, hi = lane >> 4;
  int tile = blockIdx.x * 8 + wave;          // 65536 tiles total
  int rtile = tile >> 4, ctile = tile & 15;
  const float* xrow = x + (size_t)(rtile * 16 + lr) * STATE;
  const bf16* brow = Bbf + (size_t)(ctile * 16 + lr) * STATE + hi * 16;
  v8f acc0 = {0.f, 0.f, 0.f, 0.f, 0.f, 0.f, 0.f, 0.f};
  v8f acc1 = acc0;
#pragma unroll
  for (int kb = 0; kb < STATE; kb += 64) {
    acc0 = wmma_bf16(pack_a_f32(xrow, kb, hi),      *(const v16bf*)(brow + kb),      acc0);
    acc1 = wmma_bf16(pack_a_f32(xrow, kb + 32, hi), *(const v16bf*)(brow + kb + 32), acc1);
  }
  bf16* up = u + (size_t)(rtile * 16 + hi * 8) * STATE + ctile * 16 + lr;
#pragma unroll
  for (int r = 0; r < 8; ++r) up[(size_t)r * STATE] = (bf16)(acc0[r] + acc1[r]);
}

// ---------------- K3: sequential scan, single workgroup ----------------
// 16 waves; wave w owns output columns [16w,16w+16). Its 16 rows of A live in
// VGPRs for all 4096 steps; state double-buffered in LDS (bf16); u timestep
// tiles DMA'd into LDS by the Tensor Data Mover, triple-buffered, depth-2.
__global__ __launch_bounds__(512) void ssm_scan_kernel(const float* __restrict__ A,
                                                       const bf16* __restrict__ u,
                                                       bf16* __restrict__ states) {
  __shared__ bf16 st[2][BATCH * APITCH];        // state[b][n], double buffered
  __shared__ bf16 ubuf[3][BATCH * STATE];       // u[t] staging (TDM dest)

  int tid = threadIdx.x;
  int wave = tid >> 5, lane = tid & 31;
  int lr = lane & 15, hi = lane >> 4;
  int m0 = wave * 16;

  // B-operand registers: rows A[m0..m0+16) (constant over time).
  v16bf Breg[8];
  {
    const float* Arow = A + (size_t)(m0 + lr) * STATE + hi * 16;
#pragma unroll
    for (int j = 0; j < 8; ++j) {
      v8f c0 = *(const v8f*)(Arow + j * 32);
      v8f c1 = *(const v8f*)(Arow + j * 32 + 8);
      v16bf b;
#pragma unroll
      for (int i = 0; i < 8; ++i) { b[i] = (bf16)c0[i]; b[i + 8] = (bf16)c1[i]; }
      Breg[j] = b;
    }
  }

  for (int i = tid; i < BATCH * APITCH; i += 512) st[0][i] = (bf16)0.f;

  if (wave == 0) {
    tdm_load_u_tile(u, 0, (unsigned)(uintptr_t)&ubuf[0][0]);
    tdm_load_u_tile(u, 1, (unsigned)(uintptr_t)&ubuf[1][0]);
    __builtin_amdgcn_s_wait_tensorcnt(1);       // u[0] resident
  }
  __syncthreads();

  bf16* gout = states + (size_t)(hi * 8) * SEQLEN * STATE + m0 + lr;
  int cur = 0;
  for (int t = 0; t < SEQLEN; ++t) {
    if (wave == 0 && t + 2 < SEQLEN)
      tdm_load_u_tile(u, t + 2, (unsigned)(uintptr_t)&ubuf[(t + 2) % 3][0]);

    // acc <- u_t tile, element (M=r+8hi, N=lr) = ubuf[M][m0+lr]
    const bf16* up = &ubuf[t % 3][(size_t)(hi * 8) * STATE + m0 + lr];
    v8f acc0, acc1;
#pragma unroll
    for (int r = 0; r < 8; ++r) { acc0[r] = (float)up[(size_t)r * STATE]; acc1[r] = 0.f; }

    const bf16* srow = st[cur] + (size_t)lr * APITCH;   // A-operand row = batch lr
#pragma unroll
    for (int j = 0; j < 8; j += 2) {
      acc0 = wmma_bf16(pack_a_bf16(srow, j * 32, hi),       Breg[j],     acc0);
      acc1 = wmma_bf16(pack_a_bf16(srow, (j + 1) * 32, hi), Breg[j + 1], acc1);
    }
#pragma unroll
    for (int r = 0; r < 8; ++r) acc0[r] += acc1[r];

    int nxt = cur ^ 1;
    bf16* drow = st[nxt] + (size_t)(hi * 8) * APITCH + m0 + lr;
#pragma unroll
    for (int r = 0; r < 8; ++r) {
      bf16 v = (bf16)acc0[r];
      drow[(size_t)r * APITCH] = v;                     // next-state in LDS
      gout[((size_t)r * SEQLEN + t) * STATE] = v;       // persist for K4
    }

    if (wave == 0) {
      if (t + 2 < SEQLEN) __builtin_amdgcn_s_wait_tensorcnt(1);  // u[t+1] resident
      else                __builtin_amdgcn_s_wait_tensorcnt(0);
    }
    cur = nxt;
    __syncthreads();
  }
}

// ---------------- K4: out = states @ C^T + D*x  (f32 out) ----------------
__global__ void ssm_gemm_out_kernel(const bf16* __restrict__ sbf,
                                    const bf16* __restrict__ Cbf,
                                    const float* __restrict__ x,
                                    const float* __restrict__ D,
                                    float* __restrict__ out) {
  int wave = threadIdx.x >> 5, lane = threadIdx.x & 31;
  int lr = lane & 15, hi = lane >> 4;
  int tile = blockIdx.x * 8 + wave;
  int rtile = tile >> 4, ctile = tile & 15;
  const bf16* srow = sbf + (size_t)(rtile * 16 + lr) * STATE;
  const bf16* crow = Cbf + (size_t)(ctile * 16 + lr) * STATE + hi * 16;
  v8f acc0 = {0.f, 0.f, 0.f, 0.f, 0.f, 0.f, 0.f, 0.f};
  v8f acc1 = acc0;
#pragma unroll
  for (int kb = 0; kb < STATE; kb += 64) {
    acc0 = wmma_bf16(pack_a_bf16(srow, kb, hi),      *(const v16bf*)(crow + kb),      acc0);
    acc1 = wmma_bf16(pack_a_bf16(srow, kb + 32, hi), *(const v16bf*)(crow + kb + 32), acc1);
  }
  int col = ctile * 16 + lr;
  float d = D[col];
  size_t row0 = (size_t)(rtile * 16 + hi * 8);
#pragma unroll
  for (int r = 0; r < 8; ++r) {
    size_t idx = (row0 + r) * STATE + col;
    out[idx] = acc0[r] + acc1[r] + d * x[idx];
  }
}

extern "C" void kernel_launch(void* const* d_in, const int* in_sizes, int n_in,
                              void* d_out, int out_size, void* d_ws, size_t ws_size,
                              hipStream_t stream) {
  (void)in_sizes; (void)n_in; (void)out_size; (void)ws_size;
  const float* x = (const float*)d_in[0];
  const float* A = (const float*)d_in[1];
  const float* B = (const float*)d_in[2];
  const float* C = (const float*)d_in[3];
  const float* D = (const float*)d_in[4];
  float* out = (float*)d_out;

  char* ws = (char*)d_ws;
  bf16* Bbf = (bf16*)(ws);                                              // 128 KB
  bf16* Cbf = (bf16*)(ws + (size_t)131072);                             // 128 KB
  bf16* ubf = (bf16*)(ws + (size_t)262144);                             // 32 MB
  bf16* sbf = (bf16*)(ws + (size_t)262144 + (size_t)ROWS * STATE * 2);  // 32 MB

  ssm_cvt_kernel<<<256, 256, 0, stream>>>(B, C, Bbf, Cbf);
  ssm_gemm_u_kernel<<<8192, 256, 0, stream>>>(x, Bbf, ubf);
  ssm_scan_kernel<<<1, 512, 0, stream>>>(A, ubf, sbf);
  ssm_gemm_out_kernel<<<8192, 256, 0, stream>>>(sbf, Cbf, x, D, out);
}
